// TertiaryLinear_37812892074336
// MI455X (gfx1250) — matrix operand order repaired and checked by reading, not compiled, not run on hardware
//
#include <hip/hip_runtime.h>
#include <stdint.h>

// y = x @ ternarize(W)^T ; x:(8192,2048) f32, W:(8192,2048) f32, y:(8192,8192) f32
// Pass 1: preprocess -> bf16 x_hi / x_lo (split precision) and ternary bf16 W in d_ws.
// Pass 2: bf16 WMMA GEMM, double-buffered LDS staging via GLOBAL_LOAD_ASYNC_TO_LDS_B128
//         (ASYNCcnt pipeline), v_wmma_f32_16x16x32_bf16 compute, non-temporal output
//         stores so the 256MB y stream does not evict the L2-resident operands.

typedef __attribute__((ext_vector_type(16))) __bf16 v16bf;
typedef __attribute__((ext_vector_type(8)))  __bf16 v8bf;
typedef __attribute__((ext_vector_type(4)))  __bf16 v4bf;
typedef __attribute__((ext_vector_type(8)))  float  v8f;
typedef __attribute__((ext_vector_type(4)))  float  v4f;

constexpr int Mdim = 8192;
constexpr int Ndim = 8192;
constexpr int Kdim = 2048;
constexpr size_t NELEM = (size_t)Mdim * Kdim;   // 16,777,216 (x and W have same shape)

constexpr int BM = 128;
constexpr int BN = 128;
constexpr int BK = 32;                 // = WMMA K depth
constexpr int LDP = 40;                // padded bf16 row stride (80 B: 16B-aligned, bank-conflict-free)
constexpr int TILE_EL   = BM * LDP;    // 5120 bf16 per tile
constexpr int BUF_EL    = 3 * TILE_EL; // xhi + xlo + wt per stage
constexpr int BUF_BYTES = BUF_EL * 2;  // 30720 B ; double buffered -> 61440 B dynamic LDS
constexpr int NT = Kdim / BK;          // 64 K-steps

union Frag { v16bf v; v8bf h[2]; };

// ---------------- pass 1: convert / ternarize ----------------
__global__ __launch_bounds__(256)
void preprocess(const v4f* __restrict__ X4, const v4f* __restrict__ W4,
                __bf16* __restrict__ xhi, __bf16* __restrict__ xlo,
                __bf16* __restrict__ wt)
{
  const size_t i4 = (size_t)blockIdx.x * 256 + threadIdx.x;
  const size_t e  = i4 * 4;

  // read-once f32 inputs: non-temporal so they don't displace the bf16 ws arrays in L2
  const v4f vx = __builtin_nontemporal_load(&X4[i4]);
  v4bf hv, lv;
#pragma unroll
  for (int j = 0; j < 4; ++j) {
    __bf16 h = (__bf16)vx[j];
    hv[j] = h;
    lv[j] = (__bf16)(vx[j] - (float)h);
  }
  *(v4bf*)(xhi + e) = hv;
  *(v4bf*)(xlo + e) = lv;

  const v4f vw = __builtin_nontemporal_load(&W4[i4]);
  v4bf tv;
#pragma unroll
  for (int j = 0; j < 4; ++j) {
    const float w = vw[j];
    const float t = (__builtin_fabsf(w) > 0.4f) ? (w > 0.0f ? 1.0f : -1.0f) : 0.0f;
    tv[j] = (__bf16)t;
  }
  *(v4bf*)(wt + e) = tv;
}

// ---------------- async copy helpers ----------------
__device__ __forceinline__ void async_copy16(uint32_t lds_byte, uint32_t goff_byte, uint64_t sbase)
{
  // GVS mode: mem = SADDR(64b sgpr pair) + VADDR(32b per-lane byte offset); dest = LDS[vdst]
  asm volatile("global_load_async_to_lds_b128 %0, %1, %2 offset:0"
               :: "v"(lds_byte), "v"(goff_byte), "s"(sbase) : "memory");
}

template <int N>
__device__ __forceinline__ void wait_async()
{
#if __has_builtin(__builtin_amdgcn_s_wait_asynccnt)
  __builtin_amdgcn_s_wait_asynccnt(N);
#else
  asm volatile("s_wait_asynccnt %0" :: "n"(N) : "memory");
#endif
}

// Issue one stage: 3 tiles (xhi/xlo/wt), 128x32 bf16 each; 6 async b128 per thread,
// ASYNCcnt += 6 per wave per stage.
__device__ __forceinline__ void stage(int buf, int kt, int tid,
                                      uint64_t xhiB, uint64_t xloB, uint64_t wtB)
{
  const int k0 = kt * BK;
  const uint32_t lbase = (uint32_t)buf * BUF_BYTES;
#pragma unroll
  for (int i = 0; i < 2; ++i) {
    const int row = i * 64 + (tid >> 2);       // 256 thr * 8 bf16 = 64 rows per issue
    const int col = (tid & 3) * 8;
    const uint32_t goff = (uint32_t)((row * Kdim + k0 + col) * 2);
    const uint32_t loff = (uint32_t)(row * LDP + col) * 2;
    async_copy16(lbase + loff,                   goff, xhiB);
    async_copy16(lbase + TILE_EL * 2 + loff,     goff, xloB);
    async_copy16(lbase + 2 * TILE_EL * 2 + loff, goff, wtB);
  }
}

// ---------------- pass 2: WMMA GEMM ----------------
__global__ __launch_bounds__(256)
void ternlin_wmma_async(const __bf16* __restrict__ Xhi, const __bf16* __restrict__ Xlo,
                        const __bf16* __restrict__ Wt, float* __restrict__ Y)
{
  extern __shared__ __bf16 smem[];   // 2 * BUF_EL bf16; dynamic-only => LDS offsets start at 0

  const int tid  = threadIdx.x;
  const int lane = tid & 31;
  const int wv   = tid >> 5;
  const int wm   = (wv & 3) * 32;    // 4 waves in M
  const int wn   = (wv >> 2) * 64;   // 2 waves in N
  const int r    = lane & 15;
  const int kc   = lane >> 4;

  const int bn = blockIdx.x;
  const int bm = blockIdx.y;

  const uint64_t xhiB = (uint64_t)(uintptr_t)(Xhi + (size_t)(bm * BM) * Kdim);
  const uint64_t xloB = (uint64_t)(uintptr_t)(Xlo + (size_t)(bm * BM) * Kdim);
  const uint64_t wtB  = (uint64_t)(uintptr_t)(Wt  + (size_t)(bn * BN) * Kdim);

  const v8f vzero = {0.f, 0.f, 0.f, 0.f, 0.f, 0.f, 0.f, 0.f};
  v8f acc[2][4];
#pragma unroll
  for (int i = 0; i < 2; ++i)
#pragma unroll
    for (int j = 0; j < 4; ++j) acc[i][j] = vzero;

  // prologue: fill both buffers
  stage(0, 0, tid, xhiB, xloB, wtB);
  stage(1, 1, tid, xhiB, xloB, wtB);

  for (int kt = 0; kt < NT; ++kt) {
    // wait for stage kt (allow the next stage's 6 copies to stay in flight)
    if (kt < NT - 1) wait_async<6>(); else wait_async<0>();
    __syncthreads();   // all waves' copies for this buffer are visible

    const __bf16* sb   = smem + (kt & 1) * BUF_EL;
    const __bf16* sXhi = sb;
    const __bf16* sXlo = sb + TILE_EL;
    const __bf16* sWt  = sb + 2 * TILE_EL;

    // A fragments (16-bit 16x32 layout): lane half kc -> K {kc*8..+7} then {16+kc*8..+7}
    Frag ah[2], al[2], bb[4];
#pragma unroll
    for (int mt = 0; mt < 2; ++mt) {
      const __bf16* p = &sXhi[(wm + mt * 16 + r) * LDP + kc * 8];
      ah[mt].h[0] = *(const v8bf*)p;
      ah[mt].h[1] = *(const v8bf*)(p + 16);
      const __bf16* q = &sXlo[(wm + mt * 16 + r) * LDP + kc * 8];
      al[mt].h[0] = *(const v8bf*)q;
      al[mt].h[1] = *(const v8bf*)(q + 16);
    }
    // B fragments (32x16): lane half kc -> K {kc*8..+7} then {16+kc*8..+7}
#pragma unroll
    for (int nt = 0; nt < 4; ++nt) {
      const __bf16* p = &sWt[(wn + nt * 16 + r) * LDP + kc * 8];
      bb[nt].h[0] = *(const v8bf*)p;
      bb[nt].h[1] = *(const v8bf*)(p + 16);
    }

#pragma unroll
    for (int mt = 0; mt < 2; ++mt)
#pragma unroll
      for (int nt = 0; nt < 4; ++nt) {
        acc[mt][nt] = __builtin_amdgcn_wmma_f32_16x16x32_bf16(
            false, al[mt].v, false, bb[nt].v, (short)0, acc[mt][nt], false, false);
        acc[mt][nt] = __builtin_amdgcn_wmma_f32_16x16x32_bf16(
            false, ah[mt].v, false, bb[nt].v, (short)0, acc[mt][nt], false, false);
      }

    __syncthreads();   // everyone done reading this buffer
    if (kt + 2 < NT) stage(kt & 1, kt + 2, tid, xhiB, xloB, wtB);
  }

  // epilogue: C/D layout -> row = mt*16 + kc*8 + e ; col = nt*16 + r
  // Non-temporal stores: y (256MB) is write-once; keep operands resident in L2.
#pragma unroll
  for (int mt = 0; mt < 2; ++mt)
#pragma unroll
    for (int nt = 0; nt < 4; ++nt) {
      const int col = bn * BN + wn + nt * 16 + r;
#pragma unroll
      for (int e = 0; e < 8; ++e) {
        const int row = bm * BM + wm + mt * 16 + kc * 8 + e;
        __builtin_nontemporal_store(acc[mt][nt][e], &Y[(size_t)row * Ndim + col]);
      }
    }
}

extern "C" void kernel_launch(void* const* d_in, const int* in_sizes, int n_in,
                              void* d_out, int out_size, void* d_ws, size_t ws_size,
                              hipStream_t stream) {
  (void)in_sizes; (void)n_in; (void)out_size; (void)ws_size;
  const float* X  = (const float*)d_in[0];
  const float* Wg = (const float*)d_in[1];
  float* Y = (float*)d_out;

  // workspace layout: xhi | xlo | wt  (bf16 each, 32 MB each = 96 MB total)
  __bf16* xhi = (__bf16*)d_ws;
  __bf16* xlo = xhi + NELEM;
  __bf16* wt  = xlo + NELEM;

  preprocess<<<dim3((unsigned)(NELEM / 4 / 256)), dim3(256), 0, stream>>>(
      (const v4f*)X, (const v4f*)Wg, xhi, xlo, wt);

  ternlin_wmma_async<<<dim3(Ndim / BN, Mdim / BM), dim3(256), 2 * BUF_BYTES, stream>>>(
      xhi, xlo, wt, Y);
}
